// DeepMDsimpleEnergy_858993459526
// MI455X (gfx1250) — matrix operand order repaired and checked by reading, not compiled, not run on hardware
//
#include <hip/hip_runtime.h>
#include <hip/hip_bf16.h>

// ---------------- problem constants (match reference) ----------------
#define NS      64
#define NCELLS  64
#define NP      8
#define NATOM   (NCELLS * NP)     // 512 atoms per sample
#define TOTAL_ATOMS (NS * NATOM)  // 32768
#define DDIM    32                // pyramid output width
#define DESC    (2 * DDIM)        // 64 descriptor width
#define SPLIT   8                 // neighbor-loop split factor (lanes per atom)

typedef __attribute__((ext_vector_type(16))) _Float16 v16h;
typedef __attribute__((ext_vector_type(8)))  float    v8f;

struct PyrW {
    const float* W[5];   // shapes: 1x2, 2x4, 4x8, 8x16, 16x32 (row-major d_in x d_out)
    const float* b[5];   // 2,4,8,16,32
};

// Dense + ReLU, all weight indices wave-uniform -> scalar loads.
template <int DIN, int DOUT>
__device__ __forceinline__ void dense_relu(const float* __restrict__ W,
                                           const float* __restrict__ b,
                                           const float* __restrict__ in,
                                           float* __restrict__ out) {
#pragma unroll
    for (int n = 0; n < DOUT; ++n) {
        float acc = b[n];
#pragma unroll
        for (int k = 0; k < DIN; ++k)
            acc = fmaf(in[k], W[k * DOUT + n], acc);
        out[n] = fmaxf(acc, 0.0f);
    }
}

// pyramid MLP: scalar x -> 32 outputs, ReLU after every layer
__device__ __forceinline__ void pyramid(const PyrW& p, float x, float out32[DDIM]) {
    float h2[2], h4[4], h8[8], h16[16];
#pragma unroll
    for (int n = 0; n < 2; ++n)
        h2[n] = fmaxf(fmaf(x, p.W[0][n], p.b[0][n]), 0.0f);
    dense_relu<2, 4>(p.W[1], p.b[1], h2, h4);
    dense_relu<4, 8>(p.W[2], p.b[2], h4, h8);
    dense_relu<8, 16>(p.W[3], p.b[3], h8, h16);
    dense_relu<16, 32>(p.W[4], p.b[4], h16, out32);
}

// ------------- kernel 1: per-atom descriptor (neighbor sum) -------------
// SPLIT lanes cooperate on one atom; neighbor k of the 23-list goes to lane
// part = k % 8 (t-th neighbor k = part + 8t). Partial sums are combined with
// a fixed-order 3-stage shfl_xor butterfly (deterministic), lane part==0
// writes the 64-wide descriptor row.
__global__ void k_desc(const float* __restrict__ R,
                       PyrW pyr, PyrW pyr_inv,
                       const float* __restrict__ av,
                       const float* __restrict__ stdv,
                       float* __restrict__ D) {
    int tid  = blockIdx.x * blockDim.x + threadIdx.x;
    int a    = tid >> 3;          // atom index
    int part = tid & (SPLIT - 1); // which slice of the neighbor list
    if (a >= TOTAL_ATOMS) return;

    int s   = a / NATOM;
    int rem = a % NATOM;
    int c   = rem / NP;
    int i   = rem % NP;

    const float* Rs = R + (size_t)s * NATOM;
    float x = Rs[c * NP + i];

    float av0 = av[0],  av1 = av[1];
    float sd0 = stdv[0], sd1 = stdv[1];

    float acc[DESC];
#pragma unroll
    for (int n = 0; n < DESC; ++n) acc[n] = 0.0f;

    const int self24 = NP + i;    // index of self in the 24-entry (-1,0,+1) list

#pragma unroll 1
    for (int t = 0; t < 3; ++t) {
        int k = part + SPLIT * t;           // index into 23-entry list
        if (k >= 3 * NP - 1) break;         // only part 7, t=2
        int k24 = k + (k >= self24 ? 1 : 0);
        int shp = k24 >> 3;                 // 0,1,2 -> cell c-1, c, c+1
        int j   = k24 & (NP - 1);
        int cc  = c + shp - 1;
        float off = 0.0f;
        if (cc < 0)            { cc += NCELLS; off = -(float)NCELLS; }
        else if (cc >= NCELLS) { cc -= NCELLS; off =  (float)NCELLS; }

        float y  = Rs[cc * NP + j] + off;
        float r  = fabsf(x - y);
        float g0 = (1.0f / r - av0) / sd0;  // inverse-distance channel
        float g1 = (r - av1) / sd1;         // distance channel

        float h[DDIM];
        pyramid(pyr, g1, h);                // L1 = mlp(g1) * g1
#pragma unroll
        for (int n = 0; n < DDIM; ++n) acc[n] = fmaf(h[n], g1, acc[n]);
        pyramid(pyr_inv, g0, h);            // L2 = mlp(g0) * g0
#pragma unroll
        for (int n = 0; n < DDIM; ++n) acc[DDIM + n] = fmaf(h[n], g0, acc[DDIM + n]);
    }

    // deterministic butterfly reduction over the 8 lanes of this atom
#pragma unroll
    for (int n = 0; n < DESC; ++n) {
        float v = acc[n];
        v += __shfl_xor(v, 1, 32);
        v += __shfl_xor(v, 2, 32);
        v += __shfl_xor(v, 4, 32);
        acc[n] = v;
    }

    if (part == 0) {
        float* Dp = D + (size_t)a * DESC;
#pragma unroll
        for (int n = 0; n < DESC; ++n) Dp[n] = acc[n];
    }
}

// ------------- kernel 2: fitting net, layer0 via WMMA -------------
// One wave = one 16-atom tile. GEMM: (16 x 64) @ (64 x 16) via two
// v_wmma_f32_16x16x32_f16 (K=0..31 and K=32..63), f32 accumulate.
__global__ void k_fit(const float* __restrict__ D,
                      const float* __restrict__ W0, const float* __restrict__ b0,
                      const float* __restrict__ W1, const float* __restrict__ b1,
                      const float* __restrict__ W2, const float* __restrict__ b2,
                      const float* __restrict__ W3, const float* __restrict__ b3,
                      const float* __restrict__ W4, const float* __restrict__ b4,
                      const float* __restrict__ linW, const float* __restrict__ linB,
                      float* __restrict__ F) {
    __shared__ float lds[8][16][16];           // [wave][M][N]

    const int lane = threadIdx.x & 31;
    const int wave = threadIdx.x >> 5;
    const int half = lane >> 4;                // 0: lanes 0-15, 1: lanes 16-31
    const int nl   = lane & 15;

    const int m0 = (blockIdx.x * 8 + wave) * 16;   // first atom row of tile

    // ---- A operand: 16x32 f16 tile of D (CDNA5 16-bit A layout) ----
    // lane holds M = lane&15; VGPR v (elements 2v,2v+1):
    //   v<4 : K = 8*half + 2v + bit ;  v>=4 : K = 16 + 8*half + 2(v-4) + bit
    v16h a1, a2;
    const float* __restrict__ drow = D + (size_t)(m0 + nl) * DESC;
#pragma unroll
    for (int e = 0; e < 16; ++e) {
        int v = e >> 1, bit = e & 1;
        int K = (v < 4 ? 8 * half + 2 * v : 16 + 8 * half + 2 * (v - 4)) + bit;
        a1[e] = (_Float16)drow[K];
        a2[e] = (_Float16)drow[32 + K];
    }

    // ---- B operand: 32x16 f16 tile of W0 (row K striped over lanes, N=lane&15) ----
    // element e of lane: K = 8*half + e (e<8), 16 + 8*half + (e-8) (e>=8)
    v16h bm1, bm2;
#pragma unroll
    for (int e = 0; e < 16; ++e) {
        int K = (e < 8 ? 8 * half + e : 16 + 8 * half + (e - 8));
        bm1[e] = (_Float16)W0[K * 16 + nl];
        bm2[e] = (_Float16)W0[(K + 32) * 16 + nl];
    }

    v8f cacc = {0.f, 0.f, 0.f, 0.f, 0.f, 0.f, 0.f, 0.f};
    cacc = __builtin_amdgcn_wmma_f32_16x16x32_f16(false, a1, false, bm1,
                                                  (short)0, cacc, false, false);
    cacc = __builtin_amdgcn_wmma_f32_16x16x32_f16(false, a2, false, bm2,
                                                  (short)0, cacc, false, false);

    // C/D layout: element v -> M = v + 8*half, N = lane&15
    float bias = b0[nl];
#pragma unroll
    for (int v = 0; v < 8; ++v) {
        int M = v + 8 * half;
        lds[wave][M][nl] = fmaxf(cacc[v] + bias, 0.0f);
    }
    __syncthreads();

    // ---- tail layers 16->8->4->2->1 -> linear, one atom per lane 0..15 ----
    if (lane < 16) {
        float h16[16], h8[8], h4[4], h2[2];
#pragma unroll
        for (int k = 0; k < 16; ++k) h16[k] = lds[wave][lane][k];
        dense_relu<16, 8>(W1, b1, h16, h8);
        dense_relu<8, 4>(W2, b2, h8, h4);
        dense_relu<4, 2>(W3, b3, h4, h2);
        float z = fmaxf(fmaf(h2[0], W4[0], fmaf(h2[1], W4[1], b4[0])), 0.0f);
        F[m0 + lane] = fmaf(z, linW[0], linB[0]);
    }
}

// ------------- kernel 3: deterministic per-sample energy reduction -------------
__global__ void k_energy(const float* __restrict__ F, float* __restrict__ out) {
    __shared__ float sm[256];
    int s = blockIdx.x;
    int t = threadIdx.x;
    sm[t] = F[s * NATOM + t] + F[s * NATOM + 256 + t];
    __syncthreads();
#pragma unroll
    for (int w = 128; w > 0; w >>= 1) {
        if (t < w) sm[t] += sm[t + w];
        __syncthreads();
    }
    if (t == 0) out[s] = sm[0];
}

// ------------------------------ launch ------------------------------
extern "C" void kernel_launch(void* const* d_in, const int* in_sizes, int n_in,
                              void* d_out, int out_size, void* d_ws, size_t ws_size,
                              hipStream_t stream) {
    // setup_inputs() dict order, pytree-flattened:
    // 0: R | 1..10: pyr (W,b)x5 | 11..20: pyr_inv | 21..30: fit (W,b)x5
    // 31: lin_W | 32: lin_b | 33: av | 34: std
    const float* R = (const float*)d_in[0];
    PyrW pyr, pyr_inv;
    for (int l = 0; l < 5; ++l) {
        pyr.W[l]     = (const float*)d_in[1 + 2 * l];
        pyr.b[l]     = (const float*)d_in[2 + 2 * l];
        pyr_inv.W[l] = (const float*)d_in[11 + 2 * l];
        pyr_inv.b[l] = (const float*)d_in[12 + 2 * l];
    }
    const float* fitW[5]; const float* fitB[5];
    for (int l = 0; l < 5; ++l) {
        fitW[l] = (const float*)d_in[21 + 2 * l];
        fitB[l] = (const float*)d_in[22 + 2 * l];
    }
    const float* linW = (const float*)d_in[31];
    const float* linB = (const float*)d_in[32];
    const float* av   = (const float*)d_in[33];
    const float* stdv = (const float*)d_in[34];

    float* D = (float*)d_ws;                                   // 32768*64 f32 = 8 MB
    float* F = (float*)((char*)d_ws + (size_t)TOTAL_ATOMS * DESC * sizeof(float));

    k_desc<<<(TOTAL_ATOMS * SPLIT) / 256, 256, 0, stream>>>(R, pyr, pyr_inv, av, stdv, D);
    k_fit<<<TOTAL_ATOMS / (16 * 8), 256, 0, stream>>>(D,
        fitW[0], fitB[0], fitW[1], fitB[1], fitW[2], fitB[2],
        fitW[3], fitB[3], fitW[4], fitB[4], linW, linB, F);
    k_energy<<<NS, 256, 0, stream>>>(F, (float*)d_out);
}
